// PMA_80152679678423
// MI455X (gfx1250) — matrix operand order repaired and checked by reading, not compiled, not run on hardware
//
#include <hip/hip_runtime.h>

// PMA (Pooling by Multihead Attention), B=32, N=2048, D=256, H=8, DH=32, SEEDS=32.
// Pipeline: [x -> bf16] + [Wk/Wv -> bf16] -> [Q = S@Wq^T (bf16 copy pre-scaled by
// SCALE*log2(e))] -> [fused per-(b,h) K/V projection + attention, all matmuls via
// v_wmma_f32_16x16x32_bf16, softmax row-sums via an extra ones-matrix WMMA] ->
// [O + relu(O@Wo^T), row mask, lengths_out]

#define BATCH 32
#define NKEY  2048
#define DIM   256
#define NHEAD 8
#define DHEAD 32
#define NSEED 32
#define PMA_SCALE 0.0625f                 // 1/sqrt(256)
#define QSCALE 0.09016203923f             // PMA_SCALE * log2(e): scores become exp2-ready

typedef __attribute__((ext_vector_type(16))) __bf16         v16bf;
typedef __attribute__((ext_vector_type(16))) unsigned short v16us;
typedef __attribute__((ext_vector_type(8)))  float          v8f;

union Tile16 {
  unsigned short s[16];
  uint4          q[2];
  v16us          u;
  v16bf          b;
};

union Pack8 {
  unsigned short s[8];
  uint4          q;
};

__device__ __forceinline__ unsigned short f32_to_bf16(float f) {
  union { __bf16 h; unsigned short u; } cv;
  cv.h = (__bf16)f;                       // RNE, lets the compiler emit hw cvt
  return cv.u;
}

// ---------------------------------------------------------------------------
// Kernel 0: convert x (f32) to bf16, vectorized 8 elements/thread
// ---------------------------------------------------------------------------
__global__ __launch_bounds__(256) void pma_convx_kernel(
    const float* __restrict__ x, unsigned short* __restrict__ xB)
{
  const long i = ((long)blockIdx.x * 256 + threadIdx.x) * 8;
  const float4 a = *(const float4*)(x + i);
  const float4 b = *(const float4*)(x + i + 4);
  Pack8 p;
  p.s[0] = f32_to_bf16(a.x); p.s[1] = f32_to_bf16(a.y);
  p.s[2] = f32_to_bf16(a.z); p.s[3] = f32_to_bf16(a.w);
  p.s[4] = f32_to_bf16(b.x); p.s[5] = f32_to_bf16(b.y);
  p.s[6] = f32_to_bf16(b.z); p.s[7] = f32_to_bf16(b.w);
  *(uint4*)(xB + i) = p.q;
}

// ---------------------------------------------------------------------------
// Kernel 1: convert Wk, Wv (f32, row-major 256x256) to bf16
// ---------------------------------------------------------------------------
__global__ __launch_bounds__(256) void pma_convw_kernel(
    const float* __restrict__ Wk, const float* __restrict__ Wv,
    unsigned short* __restrict__ WkB, unsigned short* __restrict__ WvB)
{
  const int i = blockIdx.x * 256 + threadIdx.x;   // grid = 256 blocks -> 65536
  WkB[i] = f32_to_bf16(Wk[i]);
  WvB[i] = f32_to_bf16(Wv[i]);
}

// ---------------------------------------------------------------------------
// Kernel 2: Q = S @ Wq^T (32 x 256); f32 copy for the residual, bf16 copy
// pre-scaled by SCALE*log2(e) so scores feed v_exp_f32 directly.
// ---------------------------------------------------------------------------
__global__ __launch_bounds__(256) void pma_q_kernel(
    const float* __restrict__ S, const float* __restrict__ Wq,
    float* __restrict__ Qf, unsigned short* __restrict__ Qb)
{
  __shared__ float srow[DIM];
  const int s = blockIdx.x;                       // 32 blocks, one seed each
  for (int i = threadIdx.x; i < DIM; i += 256) srow[i] = S[s * DIM + i];
  __syncthreads();
  const int e = threadIdx.x;
  float acc = 0.f;
  for (int d = 0; d < DIM; ++d) acc = fmaf(srow[d], Wq[e * DIM + d], acc);
  Qf[s * DIM + e] = acc;
  Qb[s * DIM + e] = f32_to_bf16(acc * QSCALE);
}

// ---------------------------------------------------------------------------
// Kernel 3: fused K/V projection + attention per (b,h).
// 256 threads = 8 waves; each wave owns 8 chunks of 32 keys (8*8*32 = 2048).
// Column (key) masking is applied ONLY at the exp stage: masked exp == 0 makes
// any unmasked K/V garbage contribute exactly zero downstream.
// ---------------------------------------------------------------------------
__global__ __launch_bounds__(256) void pma_attn_kernel(
    const unsigned short* __restrict__ xB, const int* __restrict__ lengths,
    const unsigned short* __restrict__ WkB, const unsigned short* __restrict__ WvB,
    const float* __restrict__ Qf, const unsigned short* __restrict__ Qb,
    float* __restrict__ O)
{
  __shared__ char  sRaw[49152];   // per-wave 6KB slabs, later reused for reduction
  __shared__ float sDen[NSEED];

  const int b    = blockIdx.x >> 3;
  const int h    = blockIdx.x & 7;
  const int wave = threadIdx.x >> 5;
  const int lane = threadIdx.x & 31;
  const int lm   = lane & 15;
  const int kb   = (lane >> 4) << 3;   // K base (0 / 8), unused: A tiles also use kb16
  const int kb16 = (lane >> 4) << 4;   // B-tile K base per lane half (0 / 16)
  const int mhi  = (lane >> 4) << 3;   // C-tile row offset per lane half (0 / 8)
  const int len  = lengths[b];

  unsigned short* sK  = (unsigned short*)(sRaw + wave * 6144); // [32 n][32 dh]
  unsigned short* sVt = sK + 1024;                             // [32 dh][32 n]
  unsigned short* sE  = sK + 2048;                             // [2 sh][16 m][32 n]

  // Q A-tiles (seed halves 0-15 / 16-31); K = DH = 32 fits one WMMA K-step.
  Tile16 Aq[2];
  #pragma unroll
  for (int sh = 0; sh < 2; ++sh) {
    const unsigned short* qp = Qb + (sh * 16 + lm) * DIM + h * DHEAD + kb16;
    Aq[sh].q[0] = *(const uint4*)qp;
    Aq[sh].q[1] = *(const uint4*)(qp + 8);
  }

  // all-ones B tile: rowsum(exp) = exp @ ones via WMMA (replaces scalar adds)
  Tile16 Bones;
  #pragma unroll
  for (int j = 0; j < 16; ++j) Bones.s[j] = 0x3F80;   // bf16 1.0

  const v8f zf = {0.f, 0.f, 0.f, 0.f, 0.f, 0.f, 0.f, 0.f};
  v8f accU[2][2] = {{zf, zf}, {zf, zf}};      // [seed half][dh col tile]
  v8f accRS[2]   = {zf, zf};                  // [seed half] row sums (all cols equal)

  for (int c = 0; c < 8; ++c) {
    const int n0 = ((c << 3) + wave) << 5;    // this wave's 32-key chunk
    if (c + 1 < 8)
      __builtin_prefetch(xB + ((long)b * NKEY + ((((c + 1) << 3) + wave) << 5) + lm) * DIM, 0, 1);

    // ---- fused projection: Kc/Vc (32 keys x 32 dh) = x_chunk @ W^T head slice
    v8f Kc[2][2] = {{zf, zf}, {zf, zf}};
    v8f Vc[2][2] = {{zf, zf}, {zf, zf}};
    for (int d0 = 0; d0 < DIM; d0 += 32) {
      Tile16 Ax[2];
      #pragma unroll
      for (int nh = 0; nh < 2; ++nh) {
        const unsigned short* xr =
            xB + ((long)b * NKEY + n0 + nh * 16 + lm) * DIM + d0 + kb16;
        Ax[nh].q[0] = *(const uint4*)xr;
        Ax[nh].q[1] = *(const uint4*)(xr + 8);
      }
      #pragma unroll
      for (int dc = 0; dc < 2; ++dc) {
        Tile16 Bk, Bv;
        const int wrow = (h * DHEAD + dc * 16 + lm) * DIM + d0 + kb16;
        Bk.u = *(const v16us*)(WkB + wrow);
        Bv.u = *(const v16us*)(WvB + wrow);
        #pragma unroll
        for (int nh = 0; nh < 2; ++nh) {
          Kc[nh][dc] = __builtin_amdgcn_wmma_f32_16x16x32_bf16(
              false, Ax[nh].b, false, Bk.b, (short)0, Kc[nh][dc], false, false);
          Vc[nh][dc] = __builtin_amdgcn_wmma_f32_16x16x32_bf16(
              false, Ax[nh].b, false, Bv.b, (short)0, Vc[nh][dc], false, false);
        }
      }
    }

    // ---- stage K (row-major, scattered u16) and V (transposed, packed b128)
    //      in LDS as bf16 (no masking needed here; exp mask covers it)
    #pragma unroll
    for (int nh = 0; nh < 2; ++nh)
      #pragma unroll
      for (int dc = 0; dc < 2; ++dc) {
        const int dh = dc * 16 + lm;
        Pack8 pv;
        #pragma unroll
        for (int r = 0; r < 8; ++r) {
          const int nl = nh * 16 + r + mhi;
          sK[nl * 32 + dh] = f32_to_bf16(Kc[nh][dc][r]);
          pv.s[r]          = f32_to_bf16(Vc[nh][dc][r]);
        }
        *(uint4*)(sVt + dh * 32 + nh * 16 + mhi) = pv.q;
      }

    // ---- scores = Qh @ Kc^T  (C tiles: M=seed, N=key); Q pre-scaled
    Tile16 Bs[2];
    #pragma unroll
    for (int nh = 0; nh < 2; ++nh)
      Bs[nh].u = *(const v16us*)(sK + (nh * 16 + lm) * 32 + kb16);
    v8f Sc[2][2];
    #pragma unroll
    for (int sh = 0; sh < 2; ++sh)
      #pragma unroll
      for (int nh = 0; nh < 2; ++nh)
        Sc[sh][nh] = __builtin_amdgcn_wmma_f32_16x16x32_bf16(
            false, Aq[sh].b, false, Bs[nh].b, (short)0, zf, false, false);

    // ---- raw exp2 (Q carried SCALE*log2e) + column-length mask, packed into
    //      A-tile layout for the U GEMM
    #pragma unroll
    for (int sh = 0; sh < 2; ++sh)
      #pragma unroll
      for (int nh = 0; nh < 2; ++nh) {
        const int nl = nh * 16 + lm;
        const bool ok = (n0 + nl) < len;
        #pragma unroll
        for (int r = 0; r < 8; ++r) {
          const float e = ok ? exp2f(Sc[sh][nh][r]) : 0.f;
          sE[sh * 512 + (r + mhi) * 32 + nl] = f32_to_bf16(e);
        }
      }

    // ---- U += exp @ V ; rowsum += exp @ ones
    Tile16 Ae[2], Bv2[2];
    #pragma unroll
    for (int sh = 0; sh < 2; ++sh) {
      const unsigned short* p = sE + sh * 512 + lm * 32 + kb16;
      Ae[sh].q[0] = *(const uint4*)p;
      Ae[sh].q[1] = *(const uint4*)(p + 8);
    }
    #pragma unroll
    for (int dc = 0; dc < 2; ++dc)
      Bv2[dc].u = *(const v16us*)(sVt + (dc * 16 + lm) * 32 + kb16);
    #pragma unroll
    for (int sh = 0; sh < 2; ++sh) {
      #pragma unroll
      for (int dc = 0; dc < 2; ++dc)
        accU[sh][dc] = __builtin_amdgcn_wmma_f32_16x16x32_bf16(
            false, Ae[sh].b, false, Bv2[dc].b, (short)0, accU[sh][dc], false, false);
      accRS[sh] = __builtin_amdgcn_wmma_f32_16x16x32_bf16(
          false, Ae[sh].b, false, Bones.b, (short)0, accRS[sh], false, false);
    }
  }

  // ---- deterministic cross-wave reduction (reuse the per-wave LDS slabs)
  __syncthreads();
  float* sUred = (float*)sRaw;             // [8 wave][4 tile][32 lane][8 r] = 32KB
  float* sPs   = (float*)(sRaw + 32768);   // [8 wave][2 sh][32 lane][8 r]   = 16KB
  #pragma unroll
  for (int sh = 0; sh < 2; ++sh)
    #pragma unroll
    for (int dc = 0; dc < 2; ++dc) {
      float* dst = sUred + ((wave * 4 + sh * 2 + dc) * 32 + lane) * 8;
      #pragma unroll
      for (int r = 0; r < 8; ++r) dst[r] = accU[sh][dc][r];
    }
  #pragma unroll
  for (int sh = 0; sh < 2; ++sh) {
    float* dst = sPs + ((wave * 2 + sh) * 32 + lane) * 8;
    #pragma unroll
    for (int r = 0; r < 8; ++r) dst[r] = accRS[sh][r];
  }
  __syncthreads();

  if (threadIdx.x < NSEED) {               // per-seed softmax reciprocal denom
    const int s = threadIdx.x;
    const int sh = s >> 4, rr = s & 15;
    const int r = rr & 7;
    const int lbase = (rr >= 8) ? 16 : 0;  // any column of the row-sum tile works
    float acc = 0.f;
    for (int w = 0; w < 8; ++w)
      acc += sPs[((w * 2 + sh) * 32 + lbase) * 8 + r];
    sDen[s] = 1.f / (acc + 1e-15f);
  }
  __syncthreads();

  for (int i = threadIdx.x; i < NSEED * DHEAD; i += 256) {
    const int s = i >> 5, d = i & 31;
    const int sh = s >> 4, rr = s & 15;
    const int r = rr & 7;
    const int lidx = (d & 15) + ((rr >= 8) ? 16 : 0);
    const int dc = d >> 4;
    float u = 0.f;
    for (int w = 0; w < 8; ++w)
      u += sUred[((w * 4 + sh * 2 + dc) * 32 + lidx) * 8 + r];
    O[((long)b * NSEED + s) * DIM + h * DHEAD + d] =
        Qf[s * DIM + h * DHEAD + d] + u * sDen[s];
  }
}

// ---------------------------------------------------------------------------
// Kernel 4: out = (O + relu(O @ Wo^T)) * row_mask; lengths_out = min(32, len)
// ---------------------------------------------------------------------------
__global__ __launch_bounds__(256) void pma_final_kernel(
    const float* __restrict__ O, const float* __restrict__ Wo,
    const int* __restrict__ lengths, float* __restrict__ out)
{
  __shared__ float sO[NSEED * DIM];        // 32KB: this batch's O
  const int b = blockIdx.x;
  for (int i = threadIdx.x; i < NSEED * DIM; i += 256)
    sO[i] = O[(long)b * NSEED * DIM + i];
  __syncthreads();
  const int len = lengths[b];
  const int d = threadIdx.x;               // output column
  float acc[NSEED];
  #pragma unroll
  for (int s = 0; s < NSEED; ++s) acc[s] = 0.f;
  for (int e = 0; e < DIM; ++e) {
    const float w = Wo[d * DIM + e];       // einsum("bse,de->bsd") => O @ Wo^T
    #pragma unroll
    for (int s = 0; s < NSEED; ++s) acc[s] = fmaf(sO[s * DIM + e], w, acc[s]);
  }
  for (int s = 0; s < NSEED; ++s) {
    const float y = sO[s * DIM + d] + fmaxf(acc[s], 0.f);
    out[((long)b * NSEED + s) * DIM + d] = (s < len) ? y : 0.f;
  }
  if (threadIdx.x == 0)
    out[BATCH * NSEED * DIM + b] = (float)((len < NSEED) ? len : NSEED);
}

// ---------------------------------------------------------------------------
extern "C" void kernel_launch(void* const* d_in, const int* in_sizes, int n_in,
                              void* d_out, int out_size, void* d_ws, size_t ws_size,
                              hipStream_t stream) {
  (void)in_sizes; (void)n_in; (void)out_size; (void)ws_size;
  const float* x       = (const float*)d_in[0];
  const int*   lengths = (const int*)d_in[1];
  const float* S       = (const float*)d_in[2];
  const float* Wq      = (const float*)d_in[3];
  const float* Wk      = (const float*)d_in[4];
  const float* Wv      = (const float*)d_in[5];
  const float* Wo      = (const float*)d_in[6];
  float* out = (float*)d_out;

  char* ws = (char*)d_ws;
  unsigned short* xB  = (unsigned short*)(ws);              // 32 MB
  unsigned short* WkB = (unsigned short*)(ws + 33554432);   // 128KB
  unsigned short* WvB = (unsigned short*)(ws + 33685504);   // 128KB
  float*          Qf  = (float*)(ws + 33816576);            // 32KB
  unsigned short* Qb  = (unsigned short*)(ws + 33849344);   // 16KB
  float*          O   = (float*)(ws + 33865728);            // 1MB

  pma_convx_kernel<<<8192, 256, 0, stream>>>(x, xB);
  pma_convw_kernel<<<256, 256, 0, stream>>>(Wk, Wv, WkB, WvB);
  pma_q_kernel<<<NSEED, 256, 0, stream>>>(S, Wq, Qf, Qb);
  pma_attn_kernel<<<BATCH * NHEAD, 256, 0, stream>>>(xB, lengths, WkB, WvB, Qf, Qb, O);
  pma_final_kernel<<<BATCH, 256, 0, stream>>>(O, Wo, lengths, out);
}
